// LSTMPredictor_936302870570
// MI455X (gfx1250) — matrix operand order repaired and checked by reading, not compiled, not run on hardware
//
#include <hip/hip_runtime.h>
#include <hip/hip_bf16.h>

typedef __attribute__((ext_vector_type(16))) _Float16 v16h;
typedef __attribute__((ext_vector_type(8)))  _Float16 v8h;
typedef __attribute__((ext_vector_type(8)))  float    v8f;

#define BATCH 512
#define SEQT  1024
#define HDIM  64
#define HP    72            // padded row pitch (halves) for h-state: spreads LDS banks
#define ROWS_PER_BLK 16
#define NBLK (BATCH / ROWS_PER_BLK)

// Fast device math: v_exp_f32-based exp, v_rcp_f32 reciprocal (no IEEE div sequence).
__device__ __forceinline__ float fastsig(float x){
  return __builtin_amdgcn_rcpf(1.0f + __expf(-x));
}
__device__ __forceinline__ float fasttanh(float x){
  // tanh(x) = 2*sigmoid(2x) - 1
  return __builtin_amdgcn_rcpf(1.0f + __expf(-2.0f*x)) * 2.0f - 1.0f;
}

// Build a 16x32 f16 A-operand (v16h) from an LDS-resident row-major [16][HP] f16 matrix.
// ISA layout (16-bit A 16x32): lanes 0-15 hold M=lane, K = k0 + {0..7, 16..23};
// lanes 16-31 hold M=lane-16, K = k0 + {8..15, 24..31}.
__device__ __forceinline__ v16h loadA(const _Float16* h, int row, int kb, int k0){
  const _Float16* p = h + row*HP + k0 + kb;
  v8h lo = *(const v8h*)p;        // K = k0+kb    .. +7   (16B aligned: HP*2, kb*2, k0*2 all %16==0)
  v8h hi = *(const v8h*)(p+16);   // K = k0+kb+16 .. +23
  v16h a;
#pragma unroll
  for (int i=0;i<8;++i){ a[i]=lo[i]; a[8+i]=hi[i]; }
  return a;
}

__global__ __launch_bounds__(128)
void lstm_persist(const float* __restrict__ x,
                  const float* __restrict__ Wih1, const float* __restrict__ Whh1,
                  const float* __restrict__ bih1, const float* __restrict__ bhh1,
                  const float* __restrict__ Wih2, const float* __restrict__ Whh2,
                  const float* __restrict__ bih2, const float* __restrict__ bhh2,
                  const float* __restrict__ Wl,   const float* __restrict__ bl,
                  const int* __restrict__ futp,  float* __restrict__ out)
{
  // Pre-swizzled f16 B operands: [matrix][n-tile(16)][k-chunk(2)][lane(32)][16 halves] = 96KB
  __shared__ __align__(32) _Float16 wB[3][16][2][32][16];
  __shared__ __align__(16) _Float16 h1buf[2][16*HP];   // ping-pong h1 state, f16
  __shared__ __align__(16) _Float16 h2buf[2][16*HP];   // ping-pong h2 state, f16
  __shared__ float xbuf[16];                           // per-row scalar input / fed-back output

  const int tid  = threadIdx.x;
  const int lane = tid & 31;
  const int w    = tid >> 5;          // wave id = 16-column slice of hidden dim
  const int n16  = lane & 15;
  const int hi   = (lane < 16) ? 0 : 1;
  const int b0   = blockIdx.x * ROWS_PER_BLK;
  const int FUT  = *futp;
  const int TOUT = SEQT + FUT;

  // ---- one-time: convert weights f32 -> f16 into WMMA-B lane layout ----
  // B element on lane l, half e of chunk ch for n-tile gt:  B[K][N] = W[N_global][K]
  // with N_global = gt*16 + (l&15), K = ch*32 + (l<16?0:8) + (e&7) + (e<8?0:16).
  for (int idx = tid; idx < 3*16384; idx += 128){
    int m  = idx >> 14, r = idx & 16383;
    int gt = r >> 10;  int r2 = r & 1023;
    int ch = r2 >> 9;  int r3 = r2 & 511;
    int ln = r3 >> 4;  int e  = r3 & 15;
    int n  = gt*16 + (ln & 15);
    int K  = ch*32 + ((ln < 16) ? 0 : 8) + (e & 7) + ((e & 8) ? 16 : 0);
    const float* W = (m==0) ? Whh1 : ((m==1) ? Wih2 : Whh2);   // each [256][64] row-major
    wB[m][gt][ch][ln][e] = (_Float16)W[n*64 + K];
  }
  // zero hidden state (both ping-pong buffers)
  for (int idx = tid; idx < 16*HP; idx += 128){
    h1buf[0][idx]=(_Float16)0.f; h1buf[1][idx]=(_Float16)0.f;
    h2buf[0][idx]=(_Float16)0.f; h2buf[1][idx]=(_Float16)0.f;
  }

  // Wave w owns hidden column  wcol = w*16 + n16  (lane's C-tile column).
  // Gate g's rows in the [256]-wide gate vector: n = g*64 + wcol.
  const int wcol = w*16 + n16;
  float bias1g[4], bias2g[4], wih1g[4];
#pragma unroll
  for (int g=0;g<4;++g){
    int n = g*64 + wcol;
    bias1g[g] = bih1[n] + bhh1[n];
    bias2g[g] = bih2[n] + bhh2[n];
    wih1g[g]  = Wih1[n];              // W_ih1 is [256,1]
  }
  float wl[8];
  {
    int cb = (tid & 7) * 8;
#pragma unroll
    for (int j=0;j<8;++j) wl[j] = Wl[cb+j];
  }
  const float blv = bl[0];
  float c1r[8], c2r[8];               // cell state in C-tile layout: row v+rb, col wcol
#pragma unroll
  for (int j=0;j<8;++j){ c1r[j]=0.f; c2r[j]=0.f; }

  const int arow = n16;               // A-operand row for this lane
  const int akb  = hi ? 8 : 0;        // A-operand K sub-base
  const int rb   = hi ? 8 : 0;        // C-tile row base

  __syncthreads();

  for (int tt = 0; tt < TOUT; ++tt){
    if (tt < SEQT && tid < 16) xbuf[tid] = x[(b0+tid)*SEQT + tt];
    __syncthreads();                                       // (1) x / h1[p] / h2[p] ready

    const int p = tt & 1;
    const _Float16* h1p = h1buf[p];   _Float16* h1q = h1buf[p^1];
    const _Float16* h2p = h2buf[p];   _Float16* h2q = h2buf[p^1];

    // ---- layer 1: gates + cell update, all in registers ----
    {
      v16h a0 = loadA(h1p, arow, akb, 0);
      v16h a1 = loadA(h1p, arow, akb, 32);
      float xr[8];
#pragma unroll
      for (int v=0;v<8;++v) xr[v] = xbuf[v+rb];
      v8f G[4];
#pragma unroll
      for (int g=0;g<4;++g){
        int gt = g*4 + w;
        v8f c;
#pragma unroll
        for (int v=0;v<8;++v) c[v] = __builtin_fmaf(xr[v], wih1g[g], bias1g[g]);
        v16h bB0 = *(const v16h*)&wB[0][gt][0][lane][0];
        c = __builtin_amdgcn_wmma_f32_16x16x32_f16(false,a0,false,bB0,(short)0,c,false,false);
        v16h bB1 = *(const v16h*)&wB[0][gt][1][lane][0];
        c = __builtin_amdgcn_wmma_f32_16x16x32_f16(false,a1,false,bB1,(short)0,c,false,false);
        G[g] = c;
      }
#pragma unroll
      for (int v=0;v<8;++v){
        float iv = fastsig (G[0][v]);
        float fv = fastsig (G[1][v]);
        float gv = fasttanh(G[2][v]);
        float ov = fastsig (G[3][v]);
        float cn = __builtin_fmaf(fv, c1r[v], iv*gv);
        c1r[v] = cn;
        h1q[(v+rb)*HP + wcol] = (_Float16)(ov * fasttanh(cn));
      }
    }
    __syncthreads();                                       // (2) h1[p^1] ready

    // ---- layer 2: gates + cell update ----
    {
      v16h a10 = loadA(h1q, arow, akb, 0);
      v16h a11 = loadA(h1q, arow, akb, 32);
      v16h a20 = loadA(h2p, arow, akb, 0);
      v16h a21 = loadA(h2p, arow, akb, 32);
      v8f G[4];
#pragma unroll
      for (int g=0;g<4;++g){
        int gt = g*4 + w;
        v8f c;
#pragma unroll
        for (int v=0;v<8;++v) c[v] = bias2g[g];
        v16h b0 = *(const v16h*)&wB[1][gt][0][lane][0];
        c = __builtin_amdgcn_wmma_f32_16x16x32_f16(false,a10,false,b0,(short)0,c,false,false);
        v16h b1 = *(const v16h*)&wB[1][gt][1][lane][0];
        c = __builtin_amdgcn_wmma_f32_16x16x32_f16(false,a11,false,b1,(short)0,c,false,false);
        v16h b2 = *(const v16h*)&wB[2][gt][0][lane][0];
        c = __builtin_amdgcn_wmma_f32_16x16x32_f16(false,a20,false,b2,(short)0,c,false,false);
        v16h b3 = *(const v16h*)&wB[2][gt][1][lane][0];
        c = __builtin_amdgcn_wmma_f32_16x16x32_f16(false,a21,false,b3,(short)0,c,false,false);
        G[g] = c;
      }
#pragma unroll
      for (int v=0;v<8;++v){
        float iv = fastsig (G[0][v]);
        float fv = fastsig (G[1][v]);
        float gv = fasttanh(G[2][v]);
        float ov = fastsig (G[3][v]);
        float cn = __builtin_fmaf(fv, c2r[v], iv*gv);
        c2r[v] = cn;
        h2q[(v+rb)*HP + wcol] = (_Float16)(ov * fasttanh(cn));
      }
    }
    __syncthreads();                                       // (3) h2[p^1] ready

    // ---- linear head (+ feedback in future phase) ----
    {
      int r = tid >> 3, cb = (tid & 7)*8;
      const _Float16* hp = h2q + r*HP + cb;
      v8h hv = *(const v8h*)hp;                            // one 16B LDS read
      float pacc = 0.f;
#pragma unroll
      for (int j=0;j<8;++j) pacc = __builtin_fmaf((float)hv[j], wl[j], pacc);
      pacc += __shfl_xor(pacc, 1, 8);
      pacc += __shfl_xor(pacc, 2, 8);
      pacc += __shfl_xor(pacc, 4, 8);
      if ((tid & 7) == 0){
        float ov = pacc + blv;
        out[(size_t)(b0 + r)*TOUT + tt] = ov;
        if (tt >= SEQT-1 && tt < TOUT-1) xbuf[r] = ov;     // autoregressive feedback
      }
    }
    // no barrier here: loop-top barrier (1) orders xbuf/h reads of next step
  }
}

extern "C" void kernel_launch(void* const* d_in, const int* in_sizes, int n_in,
                              void* d_out, int out_size, void* d_ws, size_t ws_size,
                              hipStream_t stream) {
  const float* x    = (const float*)d_in[0];
  const float* Wih1 = (const float*)d_in[1];
  const float* Whh1 = (const float*)d_in[2];
  const float* bih1 = (const float*)d_in[3];
  const float* bhh1 = (const float*)d_in[4];
  const float* Wih2 = (const float*)d_in[5];
  const float* Whh2 = (const float*)d_in[6];
  const float* bih2 = (const float*)d_in[7];
  const float* bhh2 = (const float*)d_in[8];
  const float* Wl   = (const float*)d_in[9];
  const float* bl   = (const float*)d_in[10];
  const int*   fut  = (const int*)d_in[11];
  (void)in_sizes; (void)n_in; (void)out_size; (void)d_ws; (void)ws_size;
  lstm_persist<<<NBLK, 128, 0, stream>>>(x, Wih1, Whh1, bih1, bhh1,
                                         Wih2, Whh2, bih2, bhh2,
                                         Wl, bl, fut, (float*)d_out);
}
